// HieraBlock_17961553232053
// MI455X (gfx1250) — matrix operand order repaired and checked by reading, not compiled, not run on hardware
//
#include <hip/hip_runtime.h>
#include <cstdint>

// ---------------------------------------------------------------------------
// Hiera block on MI455X (gfx1250): all GEMMs via v_wmma_f32_16x16x32_bf16,
// attention K/V staged into LDS with global_load_async_to_lds_b128 (ASYNCcnt).
// B=32 T=3136 DIN=192 DOUT=384 H=6 HD=64 QS=4 WS=16 INNER=64 NW=49 MLP=1536
// ---------------------------------------------------------------------------

typedef __bf16 bf16_t;
typedef __attribute__((ext_vector_type(16))) __bf16 v16bf;
typedef __attribute__((ext_vector_type(8)))  __bf16 v8bf;
typedef __attribute__((ext_vector_type(8)))  float  v8f;

#define SHUF16(a, b) __builtin_shufflevector((a), (b), 0,1,2,3,4,5,6,7,8,9,10,11,12,13,14,15)

static constexpr int kB      = 32;
static constexpr int kT      = 3136;
static constexpr int kDIN    = 192;
static constexpr int kDOUT   = 384;
static constexpr int kH      = 6;
static constexpr int kHD     = 64;
static constexpr int kWS     = 16;
static constexpr int kINNER  = 64;
static constexpr int kNW     = 49;
static constexpr int kMLP    = 1536;
static constexpr int kN1     = kB * kT;        // 100352 rows (pre-pool)
static constexpr int kTP     = kT / 4;         // 784 tokens post-pool
static constexpr int kN2     = kB * kTP;       // 25088 rows (post-pool)

// Epilogue modes for the generic WMMA GEMM
enum { EPI_BF16 = 0, EPI_GELU = 1, EPI_RESMAX = 2, EPI_ADDF32 = 3 };

// ---------------------------------------------------------------------------
// fp32 -> bf16 elementwise convert (weights)
// ---------------------------------------------------------------------------
__global__ __launch_bounds__(256) void f32_to_bf16_kernel(
    const float* __restrict__ in, bf16_t* __restrict__ out, int n) {
  int i = blockIdx.x * 256 + threadIdx.x;
  if (i < n) out[i] = (bf16_t)in[i];
}

// ---------------------------------------------------------------------------
// LayerNorm: one wave (32 lanes) per row, output bf16
// ---------------------------------------------------------------------------
__global__ __launch_bounds__(256) void layernorm_bf16_kernel(
    const float* __restrict__ x, const float* __restrict__ w,
    const float* __restrict__ b, bf16_t* __restrict__ out,
    int ncols, int nrows) {
  int row  = blockIdx.x * 8 + (threadIdx.x >> 5);
  int lane = threadIdx.x & 31;
  if (row >= nrows) return;
  const float* xr = x + (size_t)row * ncols;
  const int per = ncols >> 5;             // 6 (DIN=192) or 12 (DOUT=384)
  float vals[12];
  float s = 0.f;
#pragma unroll
  for (int i = 0; i < 12; ++i) {
    if (i < per) { vals[i] = xr[lane + 32 * i]; s += vals[i]; }
  }
#pragma unroll
  for (int off = 16; off > 0; off >>= 1) s += __shfl_xor(s, off, 32);
  float mu = s / (float)ncols;
  float vsum = 0.f;
#pragma unroll
  for (int i = 0; i < 12; ++i) {
    if (i < per) { float d = vals[i] - mu; vsum += d * d; }
  }
#pragma unroll
  for (int off = 16; off > 0; off >>= 1) vsum += __shfl_xor(vsum, off, 32);
  float inv = rsqrtf(vsum / (float)ncols + 1e-5f);
  bf16_t* orow = out + (size_t)row * ncols;
#pragma unroll
  for (int i = 0; i < 12; ++i) {
    if (i < per) {
      int c = lane + 32 * i;
      orow[c] = (bf16_t)((vals[i] - mu) * inv * w[c] + b[c]);
    }
  }
}

// ---------------------------------------------------------------------------
// Generic GEMM: C[M,N] = A[M,K](bf16) * W[N,K]^T(bf16) + bias, fused epilogue.
// Block = 256 threads = 8 waves; wave -> one 16x16 tile; block tile = 32x64.
// Fragment layouts per CDNA5 ISA (CDNA5 ISA):
//   A lane<16: K {0..7,16..23}; lane>=16: K {8..15,24..31}  (row = lane&15)
//   B lane<16: K {0..7,16..23}; lane>=16: K {8..15,24..31}  (col = lane&15)
//   D vgpr r : row m0+r (lanes 0-15) / m0+r+8 (lanes 16-31), col n0+(lane&15)
// ---------------------------------------------------------------------------
template <int EPI>
__global__ __launch_bounds__(256) void gemm_bf16_wmma_kernel(
    const bf16_t* __restrict__ A, const bf16_t* __restrict__ W,
    const float* __restrict__ bias, void* __restrict__ outp,
    const void* __restrict__ aux, int M, int N, int K) {
  const int lane  = threadIdx.x & 31;
  const int wave  = threadIdx.x >> 5;
  const int m0    = blockIdx.y * 32 + (wave >> 2) * 16;
  const int n0    = blockIdx.x * 64 + (wave & 3) * 16;
  const int akoff = (lane < 16) ? 0 : 8;
  const int bkoff = (lane < 16) ? 0 : 8;
  const bf16_t* Aptr = A + (size_t)(m0 + (lane & 15)) * K + akoff;
  const bf16_t* Wptr = W + (size_t)(n0 + (lane & 15)) * K + bkoff;

  v8f acc = {};
  for (int k0 = 0; k0 < K; k0 += 32) {
    __builtin_prefetch(Aptr + k0 + 128, 0, 1);   // global_prefetch_b8
    v8bf a0 = *(const v8bf*)(Aptr + k0);
    v8bf a1 = *(const v8bf*)(Aptr + k0 + 16);
    v8bf b0 = *(const v8bf*)(Wptr + k0);
    v8bf b1 = *(const v8bf*)(Wptr + k0 + 16);
    v16bf av = SHUF16(a0, a1);
    v16bf bv = SHUF16(b0, b1);
    acc = __builtin_amdgcn_wmma_f32_16x16x32_bf16(
        false, av, false, bv, (short)0, acc, false, false);
  }

  const int col   = n0 + (lane & 15);
  const int rbase = m0 + ((lane < 16) ? 0 : 8);
  const float bc  = bias[col];
#pragma unroll
  for (int r = 0; r < 8; ++r) {
    const int row = rbase + r;
    float v = acc[r] + bc;
    if (EPI == EPI_BF16) {
      ((bf16_t*)outp)[(size_t)row * N + col] = (bf16_t)v;
    } else if (EPI == EPI_GELU) {
      float g = 0.5f * v * (1.f + erff(v * 0.70710678118654752f));
      ((bf16_t*)outp)[(size_t)row * N + col] = (bf16_t)g;
    } else if (EPI == EPI_RESMAX) {
      // emb = attn_proj_out + max-pool(res_full) over Hiera stride-784 groups
      const bf16_t* rf = (const bf16_t*)aux;
      int bidx = row / kTP, s = row % kTP;
      size_t base = ((size_t)bidx * kT + s) * (size_t)kDOUT + col;
      float mx = (float)rf[base];
      mx = fmaxf(mx, (float)rf[base + (size_t)1 * kTP * kDOUT]);
      mx = fmaxf(mx, (float)rf[base + (size_t)2 * kTP * kDOUT]);
      mx = fmaxf(mx, (float)rf[base + (size_t)3 * kTP * kDOUT]);
      ((float*)outp)[(size_t)row * N + col] = v + mx;
    } else {  // EPI_ADDF32: final residual add -> fp32 out
      const float* e = (const float*)aux;
      ((float*)outp)[(size_t)row * N + col] = v + e[(size_t)row * N + col];
    }
  }
}

// ---------------------------------------------------------------------------
// Mask-unit attention: one wave per (b, h, window). Window = 64 tokens with
// token stride NW inside the batch; q is max-pooled 64->16 (stride-16 groups).
// K/V tiles staged global->LDS with global_load_async_to_lds_b128 (ASYNCcnt),
// WMMA for scores (16x64, K=64) and for PV (16x64, K=64). Softmax rows are
// written to d_out's `attn` tuple slot in fp32.
// ---------------------------------------------------------------------------
__global__ __launch_bounds__(32) void attention_kernel(
    const bf16_t* __restrict__ qkv, bf16_t* __restrict__ attn_out,
    float* __restrict__ attn_probs) {
  const int w = blockIdx.x, h = blockIdx.y, b = blockIdx.z;
  const int lane = threadIdx.x;

  __shared__ bf16_t ksh[kINNER][72];
  __shared__ bf16_t vsh[kINNER][72];
  __shared__ bf16_t qsh[kWS][72];
  __shared__ float  ssh[kWS][kINNER];
  __shared__ bf16_t ash[kWS][72];

  // ---- async-stage K, V into LDS (64 rows x 64 bf16 each) ----
  // One b128 async op moves 16B per lane straight into LDS, bypassing VGPRs;
  // 2 uniform outer iterations x 8 chunks x {K,V} = 32 wave instructions.
  for (int r = lane; r < kINNER; r += 32) {
    size_t grow = ((size_t)b * kT + (size_t)r * kNW + w) * (size_t)(3 * kDOUT)
                + (size_t)h * kHD;
    const bf16_t* kp = qkv + grow + kDOUT;
    const bf16_t* vp = qkv + grow + 2 * kDOUT;
    unsigned kd = (unsigned)(uintptr_t)&ksh[r][0];   // LDS byte offset
    unsigned vd = (unsigned)(uintptr_t)&vsh[r][0];
#pragma unroll
    for (int i = 0; i < 8; ++i) {
      asm volatile("global_load_async_to_lds_b128 %0, %1, off"
                   :: "v"(kd + i * 16),
                      "v"((unsigned long long)(uintptr_t)(kp + i * 8))
                   : "memory");
      asm volatile("global_load_async_to_lds_b128 %0, %1, off"
                   :: "v"(vd + i * 16),
                      "v"((unsigned long long)(uintptr_t)(vp + i * 8))
                   : "memory");
    }
  }
  // ---- q max-pool (rows {l, l+16, l+32, l+48}) with softmax scale,
  //      overlapped with the in-flight async copies ----
  if (lane < kWS) {
    const bf16_t* q0 = qkv + ((size_t)b * kT + (size_t)(lane +  0) * kNW + w) * (3 * kDOUT) + h * kHD;
    const bf16_t* q1 = qkv + ((size_t)b * kT + (size_t)(lane + 16) * kNW + w) * (3 * kDOUT) + h * kHD;
    const bf16_t* q2 = qkv + ((size_t)b * kT + (size_t)(lane + 32) * kNW + w) * (3 * kDOUT) + h * kHD;
    const bf16_t* q3 = qkv + ((size_t)b * kT + (size_t)(lane + 48) * kNW + w) * (3 * kDOUT) + h * kHD;
    for (int c = 0; c < kHD; ++c) {
      float m = fmaxf(fmaxf((float)q0[c], (float)q1[c]),
                      fmaxf((float)q2[c], (float)q3[c]));
      qsh[lane][c] = (bf16_t)(m * 0.125f);  // HD^-0.5
    }
  }
  asm volatile("s_wait_asynccnt 0x0" ::: "memory");  // K/V resident in LDS
  __syncthreads();

  const int arow  = lane & 15;
  const int akoff = (lane < 16) ? 0 : 8;
  const int bkoff = (lane < 16) ? 0 : 8;

  // ---- scores = (q*scale) @ K^T : four 16x16 tiles over 64 tokens ----
#pragma unroll
  for (int t = 0; t < 4; ++t) {
    v8f acc = {};
#pragma unroll
    for (int kk = 0; kk < kHD; kk += 32) {
      v8bf a0 = *(const v8bf*)&qsh[arow][kk + akoff];
      v8bf a1 = *(const v8bf*)&qsh[arow][kk + akoff + 16];
      const int tok = t * 16 + (lane & 15);
      v8bf b0 = *(const v8bf*)&ksh[tok][kk + bkoff];
      v8bf b1 = *(const v8bf*)&ksh[tok][kk + bkoff + 16];
      acc = __builtin_amdgcn_wmma_f32_16x16x32_bf16(
          false, SHUF16(a0, a1), false, SHUF16(b0, b1), (short)0, acc,
          false, false);
    }
    const int rb = (lane < 16) ? 0 : 8;
#pragma unroll
    for (int r = 0; r < 8; ++r) ssh[rb + r][t * 16 + (lane & 15)] = acc[r];
  }
  __syncthreads();

  // ---- row softmax (lanes 0-15, one row each); emit attn tuple output ----
  if (lane < kWS) {
    float mx = -3.0e38f;
    for (int c = 0; c < kINNER; ++c) mx = fmaxf(mx, ssh[lane][c]);
    float sum = 0.f;
    for (int c = 0; c < kINNER; ++c) sum += expf(ssh[lane][c] - mx);
    float inv = 1.f / sum;
    size_t obase =
        ((((size_t)b * kH + h) * kNW + w) * kWS + lane) * (size_t)kINNER;
    for (int c = 0; c < kINNER; ++c) {
      float p = expf(ssh[lane][c] - mx) * inv;
      attn_probs[obase + c] = p;
      ash[lane][c] = (bf16_t)p;
    }
  }
  __syncthreads();

  // ---- out = attn @ V : four 16x16 tiles over HD ----
#pragma unroll
  for (int t = 0; t < 4; ++t) {
    v8f acc = {};
#pragma unroll
    for (int kk = 0; kk < kINNER; kk += 32) {
      v8bf a0 = *(const v8bf*)&ash[arow][kk + akoff];
      v8bf a1 = *(const v8bf*)&ash[arow][kk + akoff + 16];
      const int hd = t * 16 + (lane & 15);
      const int kb = kk + bkoff;
      v16bf bv{};
#pragma unroll
      for (int i = 0; i < 16; ++i) bv[i] = vsh[kb + (i < 8 ? i : i + 8)][hd];
      acc = __builtin_amdgcn_wmma_f32_16x16x32_bf16(
          false, SHUF16(a0, a1), false, bv, (short)0, acc, false, false);
    }
    const int rb = (lane < 16) ? 0 : 8;
    const int col = h * kHD + t * 16 + (lane & 15);
#pragma unroll
    for (int r = 0; r < 8; ++r) {
      const int q = rb + r;                         // pooled token in window
      const size_t orow = (size_t)b * (kWS * kNW) + (size_t)q * kNW + w;
      attn_out[orow * (size_t)kDOUT + col] = (bf16_t)acc[r];
    }
  }
}

// ---------------------------------------------------------------------------
// Host-side orchestration
// ---------------------------------------------------------------------------
extern "C" void kernel_launch(void* const* d_in, const int* in_sizes, int n_in,
                              void* d_out, int out_size, void* d_ws,
                              size_t ws_size, hipStream_t stream) {
  const float* embeddings  = (const float*)d_in[0];
  const float* norm1_w     = (const float*)d_in[1];
  const float* norm1_b     = (const float*)d_in[2];
  const float* w_qkv       = (const float*)d_in[3];
  const float* b_qkv       = (const float*)d_in[4];
  const float* w_attn_proj = (const float*)d_in[5];
  const float* b_attn_proj = (const float*)d_in[6];
  const float* w_res_proj  = (const float*)d_in[7];
  const float* b_res_proj  = (const float*)d_in[8];
  const float* norm2_w     = (const float*)d_in[9];
  const float* norm2_b     = (const float*)d_in[10];
  const float* w_fc1       = (const float*)d_in[11];
  const float* b_fc1       = (const float*)d_in[12];
  const float* w_fc2       = (const float*)d_in[13];
  const float* b_fc2       = (const float*)d_in[14];

  char* ws = (char*)d_ws;
  size_t off = 0;
  auto alloc = [&](size_t bytes) -> char* {
    char* p = ws + off;
    off = (off + bytes + 255) & ~(size_t)255;
    return p;
  };

  bf16_t* xn       = (bf16_t*)alloc((size_t)kN1 * kDIN * 2);       //  38.5 MB
  bf16_t* qkvb     = (bf16_t*)alloc((size_t)kN1 * 3 * kDOUT * 2);  // 231.2 MB
  bf16_t* res_full = (bf16_t*)alloc((size_t)kN1 * kDOUT * 2);      //  77.1 MB
  bf16_t* attn_out = (bf16_t*)alloc((size_t)kN2 * kDOUT * 2);      //  19.3 MB
  float*  emb      = (float*) alloc((size_t)kN2 * kDOUT * 4);      //  38.5 MB
  bf16_t* h2n      = (bf16_t*)alloc((size_t)kN2 * kDOUT * 2);      //  19.3 MB
  bf16_t* hmlp     = (bf16_t*)alloc((size_t)kN2 * kMLP * 2);       //  77.1 MB
  bf16_t* wqkv_b   = (bf16_t*)alloc((size_t)3 * kDOUT * kDIN * 2);
  bf16_t* wres_b   = (bf16_t*)alloc((size_t)kDOUT * kDIN * 2);
  bf16_t* wattn_b  = (bf16_t*)alloc((size_t)kDOUT * kDOUT * 2);
  bf16_t* wfc1_b   = (bf16_t*)alloc((size_t)kMLP * kDOUT * 2);
  bf16_t* wfc2_b   = (bf16_t*)alloc((size_t)kDOUT * kMLP * 2);

  float* out_emb  = (float*)d_out;                                  // [N2,384]
  float* out_attn = (float*)d_out + (size_t)kN2 * kDOUT;            // [B,H,NW,16,64]

  auto cvt = [&](const float* src, bf16_t* dst, int n) {
    f32_to_bf16_kernel<<<(n + 255) / 256, 256, 0, stream>>>(src, dst, n);
  };
  cvt(w_qkv, wqkv_b, 3 * kDOUT * kDIN);
  cvt(w_res_proj, wres_b, kDOUT * kDIN);
  cvt(w_attn_proj, wattn_b, kDOUT * kDOUT);
  cvt(w_fc1, wfc1_b, kMLP * kDOUT);
  cvt(w_fc2, wfc2_b, kDOUT * kMLP);

  // LN1: [N1,192] -> bf16
  layernorm_bf16_kernel<<<kN1 / 8, 256, 0, stream>>>(
      embeddings, norm1_w, norm1_b, xn, kDIN, kN1);

  // qkv = xn @ w_qkv^T + b_qkv  -> bf16 [N1,1152]
  gemm_bf16_wmma_kernel<EPI_BF16>
      <<<dim3((3 * kDOUT) / 64, kN1 / 32), 256, 0, stream>>>(
          xn, wqkv_b, b_qkv, qkvb, nullptr, kN1, 3 * kDOUT, kDIN);

  // res_full = xn @ w_res^T + b_res -> bf16 [N1,384]
  gemm_bf16_wmma_kernel<EPI_BF16>
      <<<dim3(kDOUT / 64, kN1 / 32), 256, 0, stream>>>(
          xn, wres_b, b_res_proj, res_full, nullptr, kN1, kDOUT, kDIN);

  // mask-unit attention (writes attn tuple output + bf16 attn_out)
  attention_kernel<<<dim3(kNW, kH, kB), 32, 0, stream>>>(qkvb, attn_out,
                                                         out_attn);

  // emb = attn_out @ w_attn^T + b_attn + maxpool(res_full) -> fp32 [N2,384]
  gemm_bf16_wmma_kernel<EPI_RESMAX>
      <<<dim3(kDOUT / 64, kN2 / 32), 256, 0, stream>>>(
          attn_out, wattn_b, b_attn_proj, emb, res_full, kN2, kDOUT, kDOUT);

  // LN2: [N2,384] -> bf16
  layernorm_bf16_kernel<<<kN2 / 8, 256, 0, stream>>>(emb, norm2_w, norm2_b,
                                                     h2n, kDOUT, kN2);

  // hmlp = gelu(h2n @ w_fc1^T + b_fc1) -> bf16 [N2,1536]
  gemm_bf16_wmma_kernel<EPI_GELU>
      <<<dim3(kMLP / 64, kN2 / 32), 256, 0, stream>>>(
          h2n, wfc1_b, b_fc1, hmlp, nullptr, kN2, kMLP, kDOUT);

  // out_emb = emb + hmlp @ w_fc2^T + b_fc2 -> fp32 [N2,384] (tuple output 0)
  gemm_bf16_wmma_kernel<EPI_ADDF32>
      <<<dim3(kDOUT / 64, kN2 / 32), 256, 0, stream>>>(
          hmlp, wfc2_b, b_fc2, out_emb, emb, kN2, kDOUT, kMLP);
}
